// GraphDecoder_44547400794144
// MI455X (gfx1250) — matrix operand order repaired and checked by reading, not compiled, not run on hardware
//
#include <hip/hip_runtime.h>
#include <hip/hip_bf16.h>

// ---------------------------------------------------------------------------
// Geometry / model constants (match reference)
// ---------------------------------------------------------------------------
#define RG     128
#define CG     256
#define NVERT  (RG*CG)        // 32768
#define BATCH  2
#define MTOT   (BATCH*NVERT)  // 65536 rows for all GEMMs

typedef unsigned short u16;
typedef __attribute__((ext_vector_type(16))) __bf16 v16bf;
typedef __attribute__((ext_vector_type(8)))  float  v8f;

__device__ __forceinline__ float bf2f(u16 h) {
  unsigned int u = ((unsigned int)h) << 16;
  float f; __builtin_memcpy(&f, &u, 4); return f;
}
__device__ __forceinline__ u16 f2bf(float f) {
  unsigned int u; __builtin_memcpy(&u, &f, 4);
  u += 0x7fffu + ((u >> 16) & 1u);           // round-to-nearest-even
  return (u16)(u >> 16);
}

// ---------------------------------------------------------------------------
// Weight prep: Wcat[co][k] = k<cin ? W0[co][k] : W1[co][k-cin]  (bf16, padded)
// ---------------------------------------------------------------------------
__global__ void k_prep_w(const float* __restrict__ W0, const float* __restrict__ W1,
                         u16* __restrict__ out, int cout, int cin, int cout_pad) {
  int K2 = 2 * cin;
  int idx = blockIdx.x * blockDim.x + threadIdx.x;
  if (idx >= cout_pad * K2) return;
  int co = idx / K2, ci = idx % K2;
  float v = 0.f;
  if (co < cout) {
    if (ci < cin)       v = W0[co * cin + ci];
    else if (W1)        v = W1[co * cin + (ci - cin)];
  }
  out[idx] = f2bf(v);
}

// ---------------------------------------------------------------------------
// Sphere vertices (both batches)
// ---------------------------------------------------------------------------
__global__ void k_verts(float* __restrict__ vbuf) {
  int n = blockIdx.x * blockDim.x + threadIdx.x;
  if (n >= NVERT) return;
  int r = n >> 8, c = n & (CG - 1);
  float th = 0.001f + (3.14159265358979f - 0.002f) * ((float)r / (float)(RG - 1));
  float ph = 6.283185307179586f * ((float)c / (float)CG);
  float st = __sinf(th);
  float x = st * __cosf(ph), y = st * __sinf(ph), z = __cosf(th);
  float inv = rsqrtf(x * x + y * y + z * z);
  x *= inv; y *= inv; z *= inv;
  vbuf[(size_t)n * 3 + 0] = x; vbuf[(size_t)n * 3 + 1] = y; vbuf[(size_t)n * 3 + 2] = z;
  vbuf[(size_t)(NVERT + n) * 3 + 0] = x;
  vbuf[(size_t)(NVERT + n) * 3 + 1] = y;
  vbuf[(size_t)(NVERT + n) * 3 + 2] = z;
}

// ---------------------------------------------------------------------------
// First graph-conv (K=3): pure VALU, writes feats into bufX (stride ldx)
// ---------------------------------------------------------------------------
__global__ void k_first_gc(const float* __restrict__ vbuf,
                           const float* __restrict__ W0, const float* __restrict__ W1,
                           const float* __restrict__ bias,
                           u16* __restrict__ dst, int ldx) {
  int bn = blockIdx.x * blockDim.x + threadIdx.x;
  if (bn >= MTOT) return;
  int n = bn & (NVERT - 1);
  int r = n >> 8, c = n & (CG - 1);
  float p0 = vbuf[(size_t)bn * 3], p1 = vbuf[(size_t)bn * 3 + 1], p2 = vbuf[(size_t)bn * 3 + 2];
  bool lf = c > 0, rt = c < CG - 1, up = r > 0, dn = r < RG - 1;
  int offs[6]; int no = 0;
  if (lf) offs[no++] = -1;
  if (rt) offs[no++] = 1;
  if (up) offs[no++] = -CG;
  if (dn) offs[no++] = CG;
  if (up && lf) offs[no++] = -(CG + 1);
  if (dn && rt) offs[no++] = (CG + 1);
  float a0 = 0.f, a1 = 0.f, a2 = 0.f;
  for (int i = 0; i < no; ++i) {
    size_t nb = (size_t)(bn + offs[i]) * 3;
    a0 += vbuf[nb]; a1 += vbuf[nb + 1]; a2 += vbuf[nb + 2];
  }
  float inv = 1.f / (float)(no < 1 ? 1 : no);
  a0 *= inv; a1 *= inv; a2 *= inv;
  u16* o = dst + (size_t)bn * ldx;
  for (int co = 0; co < 256; ++co) {
    float s = bias[co]
            + W0[co * 3] * p0 + W0[co * 3 + 1] * p1 + W0[co * 3 + 2] * p2
            + W1[co * 3] * a0 + W1[co * 3 + 1] * a1 + W1[co * 3 + 2] * a2;
    o[co] = f2bf(s);
  }
}

// ---------------------------------------------------------------------------
// Stencil: xz[:, :C] = x ; xz[:, C:2C] = (sum of 6-neighbor x) / deg
// One block per (b,n); branch conditions are uniform per block.
// ---------------------------------------------------------------------------
__global__ __launch_bounds__(128) void k_stencil(const u16* __restrict__ x,
                                                 u16* __restrict__ xz, int C) {
  int bn = blockIdx.x;
  int n = bn & (NVERT - 1);
  int r = n >> 8, c = n & (CG - 1);
  bool lf = c > 0, rt = c < CG - 1, up = r > 0, dn = r < RG - 1;
  int deg = (int)lf + (int)rt + (int)up + (int)dn + (int)(up && lf) + (int)(dn && rt);
  if (deg < 1) deg = 1;
  float inv = 1.f / (float)deg;
  const u16* xr = x + (size_t)bn * C;
  u16* zr = xz + (size_t)bn * (size_t)(2 * C);
  long oL = -(long)C, oR = C, oU = -(long)C * CG, oD = (long)C * CG;
  for (int ch = threadIdx.x; ch < C; ch += blockDim.x) {
    float s = 0.f;
    if (lf)       s += bf2f(xr[ch + oL]);
    if (rt)       s += bf2f(xr[ch + oR]);
    if (up)       s += bf2f(xr[ch + oU]);
    if (dn)       s += bf2f(xr[ch + oD]);
    if (up && lf) s += bf2f(xr[ch + oU + oL]);
    if (dn && rt) s += bf2f(xr[ch + oD + oR]);
    zr[ch]     = xr[ch];
    zr[C + ch] = f2bf(s * inv);
  }
}

// ---------------------------------------------------------------------------
// Trilinear sample of one skip volume into bufX channels [choff, choff+C)
// ---------------------------------------------------------------------------
__global__ __launch_bounds__(128) void k_tri(const float* __restrict__ vol, int C, int R,
                                             const float* __restrict__ vbuf,
                                             u16* __restrict__ dst, int ldx, int choff) {
  int bn = blockIdx.x;
  int b = bn >> 15;  // / NVERT
  const float* pv = vbuf + (size_t)bn * 3;
  float xs = (pv[0] + 1.f) * 0.5f * (float)(R - 1);
  float ys = (pv[1] + 1.f) * 0.5f * (float)(R - 1);
  float zs = (pv[2] + 1.f) * 0.5f * (float)(R - 1);
  float fx0 = floorf(xs), fy0 = floorf(ys), fz0 = floorf(zs);
  int x0 = (int)fminf(fmaxf(fx0, 0.f), (float)(R - 2));
  int y0 = (int)fminf(fmaxf(fy0, 0.f), (float)(R - 2));
  int z0 = (int)fminf(fmaxf(fz0, 0.f), (float)(R - 2));
  float fx = fminf(fmaxf(xs - (float)x0, 0.f), 1.f);
  float fy = fminf(fmaxf(ys - (float)y0, 0.f), 1.f);
  float fz = fminf(fmaxf(zs - (float)z0, 0.f), 1.f);
  int ysp = R, zsp = R * R;
  int base = (z0 * R + y0) * R + x0;
  float w000 = (1 - fz) * (1 - fy) * (1 - fx), w001 = (1 - fz) * (1 - fy) * fx;
  float w010 = (1 - fz) * fy * (1 - fx),       w011 = (1 - fz) * fy * fx;
  float w100 = fz * (1 - fy) * (1 - fx),       w101 = fz * (1 - fy) * fx;
  float w110 = fz * fy * (1 - fx),             w111 = fz * fy * fx;
  const float* vb = vol + (size_t)b * C * zsp * R;
  u16* o = dst + (size_t)bn * ldx + choff;
  for (int ch = threadIdx.x; ch < C; ch += blockDim.x) {
    const float* vc = vb + (size_t)ch * zsp * R;
    float v = w000 * vc[base]             + w001 * vc[base + 1]
            + w010 * vc[base + ysp]       + w011 * vc[base + ysp + 1]
            + w100 * vc[base + zsp]       + w101 * vc[base + zsp + 1]
            + w110 * vc[base + zsp + ysp] + w111 * vc[base + zsp + ysp + 1];
    o[ch] = f2bf(v);
  }
}

// ---------------------------------------------------------------------------
// WMMA GEMM: out = epilogue( XZ(M x K2) @ W(cout_pad x K2)^T )
// 8 wave32s per block, each wave owns 16 rows x (NT*16) cols.
// A/B fragments loaded per the CDNA5 16-bit WMMA VGPR layouts.
// rmode: 0=none, 1=relu after add, 2=relu before add AND after add.
// ---------------------------------------------------------------------------
union FragU { v16bf v; uint4 q[2]; };

template <int NT>
__global__ __launch_bounds__(256) void k_gemm(
    const u16* __restrict__ XZ, int K2,
    const u16* __restrict__ W,
    const float* __restrict__ bias,
    const u16* __restrict__ addb, int ldadd,
    u16* __restrict__ outb, float* __restrict__ outf,
    int ldo, int cout, int rmode) {
  const int wave = threadIdx.x >> 5;
  const int lane = threadIdx.x & 31;
  const int hi   = lane >> 4;
  const int lo   = lane & 15;
  const long m0  = (long)blockIdx.x * 128 + (long)wave * 16;

  v8f zero = {0.f, 0.f, 0.f, 0.f, 0.f, 0.f, 0.f, 0.f};
  v8f acc[NT];
#pragma unroll
  for (int t = 0; t < NT; ++t) acc[t] = zero;

  // A: lane<16 holds row m0+lo, K = {kb..kb+7, kb+16..kb+23};
  //    lane>=16 same row,       K = {kb+8..15, kb+24..31}.
  const u16* arow = XZ + (size_t)(m0 + lo) * K2 + hi * 8;

  for (int kb = 0; kb < K2; kb += 32) {
    FragU a;
    a.q[0] = *(const uint4*)(arow + kb);
    a.q[1] = *(const uint4*)(arow + kb + 16);
    __builtin_prefetch(arow + kb + 128, 0, 0);   // global_prefetch path
#pragma unroll
    for (int t = 0; t < NT; ++t) {
      // B: lane = N column (lo); K = {kb+8*hi..+7, kb+16+8*hi..+7} (same as A).
      const u16* brow = W + (size_t)(t * 16 + lo) * K2 + kb + hi * 8;
      FragU b;
      b.q[0] = *(const uint4*)(brow);
      b.q[1] = *(const uint4*)(brow + 16);
      acc[t] = __builtin_amdgcn_wmma_f32_16x16x32_bf16(
          false, a.v, false, b.v, (short)0, acc[t], false, false);
    }
  }

  // C/D layout: VGPR j, lanes 0-15 -> M = m0+j; lanes 16-31 -> M = m0+8+j; N = lo.
#pragma unroll
  for (int t = 0; t < NT; ++t) {
    const int col = t * 16 + lo;
    const bool cok = (col < cout);
    const float bv = (cok && bias) ? bias[col] : 0.f;
#pragma unroll
    for (int j = 0; j < 8; ++j) {
      const long m = m0 + j + hi * 8;
      float v = acc[t][j] + bv;
      if (rmode == 2) v = v > 0.f ? v : 0.f;
      if (addb && cok) v += bf2f(addb[(size_t)m * ldadd + col]);
      if (rmode >= 1) v = v > 0.f ? v : 0.f;
      if (cok) {
        if (outb) outb[(size_t)m * ldo + col] = f2bf(v);
        if (outf) outf[(size_t)m * ldo + col] = v;
      }
    }
  }
}

// ---------------------------------------------------------------------------
// Finalize one step: emit outs[i]=[x, v+dV] (f32), deltas[i]=dV; update v.
// ---------------------------------------------------------------------------
__global__ __launch_bounds__(128) void k_finalize(const u16* __restrict__ xsrc, int ldx,
                                                  int cout, float* __restrict__ vbuf,
                                                  const float* __restrict__ dv,
                                                  float* __restrict__ outx,
                                                  float* __restrict__ outd) {
  int bn = blockIdx.x;
  const u16* xr = xsrc + (size_t)bn * ldx;
  float* o = outx + (size_t)bn * (size_t)(cout + 3);
  for (int c = threadIdx.x; c < cout; c += blockDim.x) o[c] = bf2f(xr[c]);
  if (threadIdx.x < 3) {
    int c = threadIdx.x;
    float d  = dv[(size_t)bn * 3 + c];
    float vn = vbuf[(size_t)bn * 3 + c] + d;
    vbuf[(size_t)bn * 3 + c] = vn;
    o[cout + c] = vn;
    outd[(size_t)bn * 3 + c] = d;
  }
}

// ---------------------------------------------------------------------------
// Host orchestration
// ---------------------------------------------------------------------------
static inline void launch_gemm(hipStream_t st, const u16* xz, int K2, const u16* W,
                               const float* bias, const u16* addb, int ldadd,
                               u16* outb, float* outf, int ldo, int cout, int rmode) {
  dim3 grid(MTOT / 128), block(256);
  int NT = ((cout + 15) & ~15) / 16;
  switch (NT) {
    case 1: k_gemm<1><<<grid, block, 0, st>>>(xz, K2, W, bias, addb, ldadd, outb, outf, ldo, cout, rmode); break;
    case 2: k_gemm<2><<<grid, block, 0, st>>>(xz, K2, W, bias, addb, ldadd, outb, outf, ldo, cout, rmode); break;
    case 4: k_gemm<4><<<grid, block, 0, st>>>(xz, K2, W, bias, addb, ldadd, outb, outf, ldo, cout, rmode); break;
    default: k_gemm<8><<<grid, block, 0, st>>>(xz, K2, W, bias, addb, ldadd, outb, outf, ldo, cout, rmode); break;
  }
}

struct Layer { u16* w; const float* b; int K2; int cout; };

extern "C" void kernel_launch(void* const* d_in, const int* in_sizes, int n_in,
                              void* d_out, int out_size, void* d_ws, size_t ws_size,
                              hipStream_t stream) {
  (void)in_sizes; (void)n_in; (void)out_size; (void)ws_size;

  // ---- workspace layout (bump allocator) ----
  char* wsb = (char*)d_ws;
  size_t off = 0;
  auto alloc = [&](size_t bytes) -> void* {
    void* p = wsb + off;
    off = (off + bytes + 255) & ~(size_t)255;
    return p;
  };
  float* vbuf = (float*)alloc((size_t)MTOT * 3 * 4);
  float* dv   = (float*)alloc((size_t)MTOT * 3 * 4);
  u16* bufX = (u16*)alloc((size_t)MTOT * 640 * 2);
  u16* bufB = (u16*)alloc((size_t)MTOT * 128 * 2);
  u16* bufC = (u16*)alloc((size_t)MTOT * 128 * 2);
  u16* bufD = (u16*)alloc((size_t)MTOT * 128 * 2);
  u16* XZ   = (u16*)alloc((size_t)MTOT * 1280 * 2);

  auto F = [&](int i) { return (const float*)d_in[i]; };

  // ---- weight prep (bf16 [W0|W1] concat, padded cout) ----
  auto prep = [&](const float* W0, const float* W1, int cout, int cin) -> u16* {
    int cp = (cout + 15) & ~15;
    u16* dst = (u16*)alloc((size_t)cp * 2 * cin * 2);
    int ne = cp * 2 * cin;
    k_prep_w<<<(ne + 255) / 256, 256, 0, stream>>>(W0, W1, dst, cout, cin, cp);
    return dst;
  };

  const int CIN[3]  = {640, 224, 112};
  const int COUT[3] = {128, 64, 32};
  const int AGG[3][2] = {{4, 3}, {2, 1}, {1, 0}};
  const int SKC[5] = {16, 32, 64, 128, 256};
  const int SKR[5] = {64, 32, 16, 8, 4};
  const int FEAT[3] = {256, 128, 64};  // GRAPH_CH[i]

  // param leaves: pytree flatten (sorted dict keys): first{W0,W1,b}, then steps
  int ci = 5;
  const float* fW0 = F(ci++); const float* fW1 = F(ci++); const float* fB = F(ci++);

  Layer L[3][9];  // [0]=b0.gc0 [1]=b0.proj [2]=b0.gc1 [3..6]=b1/b2 gc0/gc1 [7]=connect [8]=f2v
  for (int s = 0; s < 3; ++s) {
    int cin = CIN[s], cout = COUT[s];
    const float *w0, *w1, *bb;
    w0 = F(ci++); w1 = F(ci++); bb = F(ci++);
    L[s][0] = {prep(w0, w1, cout, cin), bb, 2 * cin, cout};
    w0 = F(ci++); w1 = F(ci++); bb = F(ci++);
    L[s][2] = {prep(w0, w1, cout, cout), bb, 2 * cout, cout};
    const float* pw = F(ci++);
    L[s][1] = {prep(pw, nullptr, cout, cin), nullptr, 2 * cin, cout};
    for (int b = 1; b < 3; ++b) {
      w0 = F(ci++); w1 = F(ci++); bb = F(ci++);
      L[s][1 + 2 * b] = {prep(w0, w1, cout, cout), bb, 2 * cout, cout};
      w0 = F(ci++); w1 = F(ci++); bb = F(ci++);
      L[s][2 + 2 * b] = {prep(w0, w1, cout, cout), bb, 2 * cout, cout};
    }
    if (s < 2) {  // 'connect' sorts before 'f2v'
      w0 = F(ci++); w1 = F(ci++); bb = F(ci++);
      L[s][7] = {prep(w0, w1, cout, cout), bb, 2 * cout, cout};
    } else {
      L[s][7] = {nullptr, nullptr, 0, 0};
    }
    w0 = F(ci++); w1 = F(ci++); bb = F(ci++);
    L[s][8] = {prep(w0, w1, 3, cout), bb, 2 * cout, 3};
  }

  // ---- output regions ----
  float* out0 = (float*)d_out;
  float* out1 = out0 + (size_t)MTOT * 131;
  float* out2 = out1 + (size_t)MTOT * 67;
  float* dd0  = out2 + (size_t)MTOT * 35;
  float* dd1  = dd0 + (size_t)MTOT * 3;
  float* dd2  = dd1 + (size_t)MTOT * 3;
  float* outs[3] = {out0, out1, out2};
  float* dels[3] = {dd0, dd1, dd2};

  // ---- pipeline ----
  k_verts<<<(NVERT + 127) / 128, 128, 0, stream>>>(vbuf);
  k_first_gc<<<(MTOT + 255) / 256, 256, 0, stream>>>(vbuf, fW0, fW1, fB, bufX, 640);

  dim3 gBN(MTOT), bST(128);
  for (int s = 0; s < 3; ++s) {
    int cin = CIN[s], cout = COUT[s];
    int i1 = AGG[s][0], i2 = AGG[s][1];
    // skip samples into bufX channels [FEAT[s], cin)
    k_tri<<<gBN, bST, 0, stream>>>(F(i1), SKC[i1], SKR[i1], vbuf, bufX, cin, FEAT[s]);
    k_tri<<<gBN, bST, 0, stream>>>(F(i2), SKC[i2], SKR[i2], vbuf, bufX, cin, FEAT[s] + SKC[i1]);

    // resblock 0: h=relu(gc0(x)); res=x@proj; out=relu(relu(gc1(h))+res)
    k_stencil<<<gBN, bST, 0, stream>>>(bufX, XZ, cin);
    launch_gemm(stream, XZ, L[s][0].K2, L[s][0].w, L[s][0].b, nullptr, 0, bufB, nullptr, cout, cout, 1);
    launch_gemm(stream, XZ, L[s][1].K2, L[s][1].w, nullptr,   nullptr, 0, bufC, nullptr, cout, cout, 0);
    k_stencil<<<gBN, bST, 0, stream>>>(bufB, XZ, cout);
    launch_gemm(stream, XZ, L[s][2].K2, L[s][2].w, L[s][2].b, bufC, cout, bufD, nullptr, cout, cout, 2);

    // resblock 1 (identity residual = bufD)
    k_stencil<<<gBN, bST, 0, stream>>>(bufD, XZ, cout);
    launch_gemm(stream, XZ, L[s][3].K2, L[s][3].w, L[s][3].b, nullptr, 0, bufB, nullptr, cout, cout, 1);
    k_stencil<<<gBN, bST, 0, stream>>>(bufB, XZ, cout);
    launch_gemm(stream, XZ, L[s][4].K2, L[s][4].w, L[s][4].b, bufD, cout, bufC, nullptr, cout, cout, 2);

    // resblock 2 (identity residual = bufC)
    k_stencil<<<gBN, bST, 0, stream>>>(bufC, XZ, cout);
    launch_gemm(stream, XZ, L[s][5].K2, L[s][5].w, L[s][5].b, nullptr, 0, bufB, nullptr, cout, cout, 1);
    k_stencil<<<gBN, bST, 0, stream>>>(bufB, XZ, cout);
    launch_gemm(stream, XZ, L[s][6].K2, L[s][6].w, L[s][6].b, bufC, cout, bufD, nullptr, cout, cout, 2);

    // shared stencil for f2v + connect
    k_stencil<<<gBN, bST, 0, stream>>>(bufD, XZ, cout);
    launch_gemm(stream, XZ, L[s][8].K2, L[s][8].w, L[s][8].b, nullptr, 0, nullptr, dv, 3, 3, 0);

    const u16* xsrc = bufD; int ldxe = cout;
    if (s < 2) {
      launch_gemm(stream, XZ, L[s][7].K2, L[s][7].w, L[s][7].b, nullptr, 0, bufX, nullptr, CIN[s + 1], cout, 1);
      xsrc = bufX; ldxe = CIN[s + 1];
    }
    k_finalize<<<gBN, bST, 0, stream>>>(xsrc, ldxe, cout, vbuf, dv, outs[s], dels[s]);
  }
}
